// FedGR_58488864637275
// MI455X (gfx1250) — matrix operand order repaired and checked by reading, not compiled, not run on hardware
//
#include <hip/hip_runtime.h>

// ---------------- problem constants ----------------
#define NN    30000
#define NNP   30016        // rows padded to multiple of 32 (no row guards in GEMM)
#define EE    240000
#define GG    1500
#define EMB   300
#define EMBP  320          // K dim padded to multiple of 32
#define HID   600
#define HIDP  608          // K dim padded to multiple of 32
#define NP1   768          // HIDP rounded up to 256-col GEMM blocks (B row padding)
#define NP2   512          // EMBP rounded up to 256-col GEMM blocks
#define NDIM  9
#define EDIM  3
#define TASKS 12

typedef __attribute__((ext_vector_type(16))) _Float16 v16h;
typedef __attribute__((ext_vector_type(8)))  float    v8f;
typedef __attribute__((ext_vector_type(4)))  _Float16 v4h;

// ======================================================================
// Generic utility kernels
// ======================================================================
__global__ void zero_f32(float* __restrict__ p, long n) {
  long i = (long)blockIdx.x * 256 + threadIdx.x;
  if (i < n) p[i] = 0.f;
}

__global__ void copy_f32(float* __restrict__ dst, const float* __restrict__ src, long n) {
  long i = (long)blockIdx.x * 256 + threadIdx.x;
  if (i < n) dst[i] = src[i];
}

// Convert + transpose f32 weights [L][Ks][Ns] (row major) -> f16 [L][NpPad][Kp]
// (N-major so GEMM B staging is contiguous; zero pad both dims)
__global__ void convert_pad_f16_t(const float* __restrict__ src, _Float16* __restrict__ dst,
                                  int L, int Ks, int Ns, int Kp, int NpPad) {
  long idx = (long)blockIdx.x * 256 + threadIdx.x;
  long total = (long)L * NpPad * Kp;
  if (idx >= total) return;
  int k = (int)(idx % Kp);
  long t = idx / Kp;
  int n = (int)(t % NpPad);
  int l = (int)(t / NpPad);
  float v = 0.f;
  if (k < Ks && n < Ns) v = src[((long)l * Ks + k) * Ns + n];
  dst[idx] = (_Float16)v;
}

// ======================================================================
// Node encoder: x_feat = x @ W_enc + b_enc; pad rows/cols = 0
// ======================================================================
__global__ void encoder_kernel(const float* __restrict__ x, const float* __restrict__ W,
                               const float* __restrict__ b,
                               float* __restrict__ xg, float* __restrict__ xr) {
  int idx = blockIdx.x * 256 + threadIdx.x;
  if (idx >= NNP * EMBP) return;
  int n = idx / EMBP, j = idx - n * EMBP;
  float v = 0.f;
  if (n < NN && j < EMB) {
    v = b[j];
    const float* xn = x + n * NDIM;
#pragma unroll
    for (int k = 0; k < NDIM; ++k) v += xn[k] * W[k * EMB + j];
  }
  xg[idx] = v;
  xr[idx] = v;
}

// ======================================================================
// Fused edge-MLP + message + scatter:  agg[dst] += relu(x[src] + edge_attr@We + be)
// ======================================================================
__global__ void scatter_kernel(const int* __restrict__ ei, const float* __restrict__ ea,
                               const float* __restrict__ We, const float* __restrict__ be,
                               const float* __restrict__ xs, float* __restrict__ agg) {
  int idx = blockIdx.x * 256 + threadIdx.x;
  if (idx >= EE * (EMB / 4)) return;
  int e = idx / (EMB / 4);
  int c = (idx - e * (EMB / 4)) * 4;
  int src = ei[e], dst = ei[EE + e];
  float a0 = ea[e * 3 + 0], a1 = ea[e * 3 + 1], a2 = ea[e * 3 + 2];
  const float* xp = xs + (size_t)src * EMBP + c;
  float* ap = agg + (size_t)dst * EMBP + c;
#pragma unroll
  for (int u = 0; u < 4; ++u) {
    int j = c + u;
    float ev = be[j] + a0 * We[j] + a1 * We[EMB + j] + a2 * We[2 * EMB + j];
    float m = xp[u] + ev;
    m = m > 0.f ? m : 0.f;
    unsafeAtomicAdd(ap + u, m);   // native global_atomic_add_f32
  }
}

// ======================================================================
// WMMA GEMM:  C = epilogue( A(f32, NNPxKp) @ B + bias ),  B given as
// pre-transposed f16 Bt[NpPad][Kp] (N-major, zero padded -> no staging guards).
// Block: 256 thr = 8 waves; tile 32(M) x 256(N); K-tile 32 via LDS.
// Fragment layouts per CDNA5 ISA 7.12.2.
// ======================================================================
__global__ __launch_bounds__(256)
void wmma_gemm(const float* __restrict__ A, int Kp,
               const _Float16* __restrict__ Bt, int Np,
               const float* __restrict__ bias, int nReal,
               const float* __restrict__ Xres,
               float* __restrict__ C, int relu_out) {
  __shared__ _Float16 As[32][40];      // [row][k]   stride 80B (8B aligned vectors)
  __shared__ _Float16 BsT[256][40];    // [col][k]

  const int tid  = threadIdx.x;
  const int lane = tid & 31;
  const int wave = tid >> 5;
  const int msub = wave & 1;           // 16-row sub-tile
  const int ngrp = wave >> 1;          // 64-col group
  const int bm = blockIdx.y * 32;
  const int bn = blockIdx.x * 256;

  // staging coordinates (all unguarded, fully coalesced)
  const int ar = tid >> 3;             // A row 0..31
  const int ac = (tid & 7) << 2;       // A col (float4)

  v8f acc[4] = {};

  const int kTiles = Kp >> 5;
  for (int kt = 0; kt < kTiles; ++kt) {
    // ---- stage A: 32x32 f32 -> f16, one float4 per thread ----
    {
      const float4 av = *(const float4*)(A + (size_t)(bm + ar) * Kp + (kt << 5) + ac);
      v4h h;
      h[0] = (_Float16)av.x; h[1] = (_Float16)av.y;
      h[2] = (_Float16)av.z; h[3] = (_Float16)av.w;
      *(v4h*)&As[ar][ac] = h;
    }
    // ---- stage B: 256 cols x 32 k halves, contiguous both sides ----
#pragma unroll
    for (int i = 0; i < 4; ++i) {
      int e = tid + (i << 8);          // 0..1023
      int col = e >> 2;                // 0..255
      int c8  = (e & 3) << 3;          // k chunk of 8 halves
      const _Float16* gp = Bt + (size_t)(bn + col) * Kp + (kt << 5) + c8;
      v4h b0 = *(const v4h*)(gp);
      v4h b1 = *(const v4h*)(gp + 4);
      *(v4h*)&BsT[col][c8]     = b0;
      *(v4h*)&BsT[col][c8 + 4] = b1;
    }
    __syncthreads();

    // ---- A fragment (ISA: lane<16 K base 0, lane>=16 K base 8; elems 8..15 at +16)
    const int arow = msub * 16 + (lane & 15);
    const int ka = (lane >> 4) << 3;
    v16h af;
#pragma unroll
    for (int i = 0; i < 8; ++i) af[i] = As[arow][ka + i];
#pragma unroll
    for (int i = 0; i < 8; ++i) af[8 + i] = As[arow][16 + ka + i];

    // ---- B fragments + 4 WMMAs (A fragment reused) ----
    const int kb = (lane >> 4) << 3;
    const int cb = ngrp * 64 + (lane & 15);
#pragma unroll
    for (int t = 0; t < 4; ++t) {
      v16h bf;
#pragma unroll
      for (int i = 0; i < 16; ++i) bf[i] = BsT[cb + t * 16][kb + (i & 7) + ((i >> 3) << 4)];
      acc[t] = __builtin_amdgcn_wmma_f32_16x16x32_f16(
          false, af, false, bf, (short)0, acc[t], false, false);
    }
    __syncthreads();
  }

  // ---- epilogue: C layout VGPR r -> M = r + 8*(lane>=16), N = lane&15 ----
  const int rowHi = (lane >> 4) << 3;
#pragma unroll
  for (int t = 0; t < 4; ++t) {
    const int gn = bn + ngrp * 64 + t * 16 + (lane & 15);
    if (gn < Np) {
#pragma unroll
      for (int r = 0; r < 8; ++r) {
        int gm = bm + msub * 16 + rowHi + r;
        float v = acc[t][r] + ((gn < nReal) ? bias[gn] : 0.f);
        if (relu_out) v = v > 0.f ? v : 0.f;
        if (Xres) v += Xres[(size_t)gm * Np + gn];
        C[(size_t)gm * Np + gn] = v;
      }
    }
  }
}

// ======================================================================
// Gate stage 2: logits = h1 @ Wg2 + bg2 + gumbel; gate = softmax[:,1]
// ======================================================================
__global__ __launch_bounds__(256)
void gate2_kernel(const float* __restrict__ h1, const float* __restrict__ Wg2,
                  const float* __restrict__ bg2, const float* __restrict__ gumbel,
                  float* __restrict__ gate) {
  __shared__ float red0[256], red1[256];
  int n = blockIdx.x, tid = threadIdx.x;
  const float* hp = h1 + (size_t)n * HIDP;
  float l0 = 0.f, l1 = 0.f;
  for (int t = tid; t < HID; t += 256) {
    float hv = hp[t];
    l0 += hv * Wg2[2 * t + 0];
    l1 += hv * Wg2[2 * t + 1];
  }
  red0[tid] = l0; red1[tid] = l1; __syncthreads();
  for (int s = 128; s > 0; s >>= 1) {
    if (tid < s) { red0[tid] += red0[tid + s]; red1[tid] += red1[tid + s]; }
    __syncthreads();
  }
  if (tid == 0) {
    float a = red0[0] + bg2[0] + gumbel[2 * n + 0];
    float b = red1[0] + bg2[1] + gumbel[2 * n + 1];
    float m = fmaxf(a, b);
    float ea = expf(a - m), eb = expf(b - m);
    gate[n] = eb / (ea + eb);
  }
}

// ======================================================================
// Graph segment reductions
// ======================================================================
__global__ void count_kernel(const int* __restrict__ batch, float* __restrict__ cnt) {
  int n = blockIdx.x * 256 + threadIdx.x;
  if (n < NN) unsafeAtomicAdd(&cnt[batch[n]], 1.f);
}

__global__ void greduce_kernel(const float* __restrict__ xg, const float* __restrict__ gate,
                               const int* __restrict__ batch,
                               float* __restrict__ hr, float* __restrict__ henv,
                               float* __restrict__ hout,
                               float* __restrict__ rnum, float* __restrict__ envnum) {
  int idx = blockIdx.x * 256 + threadIdx.x;
  if (idx >= NN * EMB) return;
  int n = idx / EMB, j = idx - n * EMB;
  int g = batch[n];
  float h  = xg[(size_t)n * EMBP + j];
  float gt = gate[n];
  unsafeAtomicAdd(&hr  [(size_t)g * EMBP + j], gt * h);
  unsafeAtomicAdd(&henv[(size_t)g * EMBP + j], (1.f - gt) * h);
  unsafeAtomicAdd(&hout[(size_t)g * EMBP + j], h);
  if (j == 0) {
    unsafeAtomicAdd(&rnum[g], gt);
    unsafeAtomicAdd(&envnum[g], 1.f - gt);
  }
}

__global__ void finalize_kernel(const float* __restrict__ cnt,
                                float* __restrict__ hr, float* __restrict__ henv,
                                float* __restrict__ hout,
                                float* __restrict__ rnum, float* __restrict__ envnum) {
  int idx = blockIdx.x * 256 + threadIdx.x;
  if (idx >= GG * EMBP) return;
  int g = idx / EMBP;
  float c = fmaxf(cnt[g], 1.f);
  hr[idx]   /= c;
  henv[idx] /= c;
  hout[idx] /= c;
  if ((idx - g * EMBP) == 0) { rnum[g] += 1e-8f; envnum[g] += 1e-8f; }
}

__global__ void norms_kernel(const float* __restrict__ hr, const float* __restrict__ hout,
                             const float* __restrict__ henv,
                             float* __restrict__ na, float* __restrict__ nb,
                             float* __restrict__ nc) {
  int g = blockIdx.x * 256 + threadIdx.x;
  if (g >= GG) return;
  float sa = 0.f, sb = 0.f, sc = 0.f;
  for (int k = 0; k < EMB; ++k) {
    float a = hr  [(size_t)g * EMBP + k];
    float b = hout[(size_t)g * EMBP + k];
    float c = henv[(size_t)g * EMBP + k];
    sa += a * a; sb += b * b; sc += c * c;
  }
  na[g] = sqrtf(sa); nb[g] = sqrtf(sb); nc[g] = sqrtf(sc);
}

// ======================================================================
// Prediction MLPs: pred_rep = mlp2(h_r + h_env[perm]); pred_rem = mlp2(h_r)
// ======================================================================
__global__ __launch_bounds__(256)
void pred_kernel(const float* __restrict__ hr, const float* __restrict__ henv,
                 const int* __restrict__ perm,
                 const float* __restrict__ Wp1, const float* __restrict__ bp1,
                 const float* __restrict__ Wp2, const float* __restrict__ bp2,
                 float* __restrict__ out) {
  __shared__ float hA[EMB], hB[EMB];
  __shared__ float sRem[TASKS], sRep[TASKS];
  int g = blockIdx.x, tid = threadIdx.x;
  int p = perm[g];
  for (int j = tid; j < EMB; j += 256) {
    float v = hr[(size_t)g * EMBP + j];
    hA[j] = v;
    hB[j] = v + henv[(size_t)p * EMBP + j];
  }
  if (tid < TASKS) { sRem[tid] = 0.f; sRep[tid] = 0.f; }
  __syncthreads();
  for (int t = tid; t < HID; t += 256) {
    float a = bp1[t], b = bp1[t];
    for (int k = 0; k < EMB; ++k) {
      float w = Wp1[k * HID + t];
      a += hA[k] * w;
      b += hB[k] * w;
    }
    a = a > 0.f ? a : 0.f;
    b = b > 0.f ? b : 0.f;
#pragma unroll
    for (int o = 0; o < TASKS; ++o) {
      atomicAdd(&sRem[o], a * Wp2[t * TASKS + o]);
      atomicAdd(&sRep[o], b * Wp2[t * TASKS + o]);
    }
  }
  __syncthreads();
  if (tid < TASKS) {
    out[(size_t)g * TASKS + tid]              = sRep[tid] + bp2[tid];   // pred_rep
    out[GG * TASKS + (size_t)g * TASKS + tid] = sRem[tid] + bp2[tid];   // pred_rem
  }
}

// ======================================================================
// Losses
// ======================================================================
__global__ __launch_bounds__(256)
void lreg_kernel(const float* __restrict__ rnum, const float* __restrict__ envnum,
                 float* __restrict__ out) {
  __shared__ float red[256];
  int tid = threadIdx.x;
  float local = 0.f;
  for (int g = tid; g < GG; g += 256) {
    float r = rnum[g], e = envnum[g];
    local += fabsf(r / (r + e) - 0.4f);
  }
  red[tid] = local; __syncthreads();
  for (int s = 128; s > 0; s >>= 1) { if (tid < s) red[tid] += red[tid + s]; __syncthreads(); }
  if (tid == 0) out[0] = red[0] / (float)GG;
}

__global__ __launch_bounds__(256)
void lcon_kernel(const float* __restrict__ hr, const float* __restrict__ henv,
                 const float* __restrict__ hout,
                 const float* __restrict__ na, const float* __restrict__ nb,
                 const float* __restrict__ nc, float* __restrict__ lcon) {
  __shared__ float hrl[EMB];
  __shared__ float red[256];
  int g = blockIdx.x, tid = threadIdx.x;
  for (int j = tid; j < EMB; j += 256) hrl[j] = hr[(size_t)g * EMBP + j];
  __syncthreads();
  float local = 0.f;
  float nag = na[g];
  for (int j = tid; j < GG; j += 256) {
    float d = 0.f;
    const float* hp = henv + (size_t)j * EMBP;
    for (int k = 0; k < EMB; ++k) d += hrl[k] * hp[k];
    local += expf(d / (nag * nc[j] + 1e-8f) / 0.2f);
  }
  red[tid] = local; __syncthreads();
  for (int s = 128; s > 0; s >>= 1) { if (tid < s) red[tid] += red[tid + s]; __syncthreads(); }
  if (tid == 0) {
    float d = 0.f;
    const float* ho = hout + (size_t)g * EMBP;
    for (int k = 0; k < EMB; ++k) d += hrl[k] * ho[k];
    float pos = expf(d / (nag * nb[g] + 1e-8f) / 0.2f);
    lcon[g] = -logf(pos / (red[0] + pos));
  }
}

__global__ __launch_bounds__(256)
void mean_kernel(const float* __restrict__ v, int n, float* __restrict__ out) {
  __shared__ float red[256];
  int tid = threadIdx.x;
  float local = 0.f;
  for (int i = tid; i < n; i += 256) local += v[i];
  red[tid] = local; __syncthreads();
  for (int s = 128; s > 0; s >>= 1) { if (tid < s) red[tid] += red[tid + s]; __syncthreads(); }
  if (tid == 0) out[0] = red[0] / (float)n;
}

// ======================================================================
// Host launcher
// ======================================================================
static inline int cdiv(long a, long b) { return (int)((a + b - 1) / b); }

static void launch_gemm(const float* A, int Kp, const _Float16* Bt, int Np,
                        const float* bias, int nReal, const float* Xres,
                        float* C, int relu_out, hipStream_t s) {
  dim3 grid(cdiv(Np, 256), NNP / 32);
  wmma_gemm<<<grid, 256, 0, s>>>(A, Kp, Bt, Np, bias, nReal, Xres, C, relu_out);
}

extern "C" void kernel_launch(void* const* d_in, const int* in_sizes, int n_in,
                              void* d_out, int out_size, void* d_ws, size_t ws_size,
                              hipStream_t stream) {
  (void)in_sizes; (void)n_in; (void)out_size; (void)ws_size;

  const float* x         = (const float*)d_in[0];
  const float* edge_attr = (const float*)d_in[1];
  const int*   edge_idx  = (const int*)  d_in[2];
  const int*   batch     = (const int*)  d_in[3];
  const float* gumbel    = (const float*)d_in[4];
  const int*   perm      = (const int*)  d_in[5];
  const float* W_enc = (const float*)d_in[6];
  const float* b_enc = (const float*)d_in[7];
  const float* We_g  = (const float*)d_in[8];
  const float* be_g  = (const float*)d_in[9];
  const float* W1_g  = (const float*)d_in[10];
  const float* b1_g  = (const float*)d_in[11];
  const float* W2_g  = (const float*)d_in[12];
  const float* b2_g  = (const float*)d_in[13];
  const float* We_r  = (const float*)d_in[14];
  const float* be_r  = (const float*)d_in[15];
  const float* W1_r  = (const float*)d_in[16];
  const float* b1_r  = (const float*)d_in[17];
  const float* W2_r  = (const float*)d_in[18];
  const float* b2_r  = (const float*)d_in[19];
  const float* Wg1   = (const float*)d_in[20];
  const float* bg1   = (const float*)d_in[21];
  const float* Wg2   = (const float*)d_in[22];
  const float* bg2   = (const float*)d_in[23];
  const float* Wp1   = (const float*)d_in[24];
  const float* bp1   = (const float*)d_in[25];
  const float* Wp2   = (const float*)d_in[26];
  const float* bp2   = (const float*)d_in[27];
  float* out = (float*)d_out;

  // -------- workspace layout --------
  char* w = (char*)d_ws;
  size_t off = 0;
  auto take = [&](size_t bytes) -> void* {
    void* p = w + off;
    off += (bytes + 255) & ~(size_t)255;
    return p;
  };
  float* xg  = (float*)take((size_t)NNP * EMBP * 4);
  float* xr  = (float*)take((size_t)NNP * EMBP * 4);
  float* agg = (float*)take((size_t)NNP * EMBP * 4);
  float* h1  = (float*)take((size_t)NNP * HIDP * 4);
  _Float16* w1g  = (_Float16*)take((size_t)5 * NP1 * EMBP * 2);   // [5][768][320]
  _Float16* w2g  = (_Float16*)take((size_t)5 * NP2 * HIDP * 2);   // [5][512][608]
  _Float16* w1r  = (_Float16*)take((size_t)2 * NP1 * EMBP * 2);
  _Float16* w2r  = (_Float16*)take((size_t)2 * NP2 * HIDP * 2);
  _Float16* wg1h = (_Float16*)take((size_t)NP1 * EMBP * 2);
  float* gate = (float*)take((size_t)NN * 4);
  // zeroed region begins here:
  float* hr     = (float*)take((size_t)GG * EMBP * 4);
  float* henv   = (float*)take((size_t)GG * EMBP * 4);
  float* hout   = (float*)take((size_t)GG * EMBP * 4);
  float* rnum   = (float*)take((size_t)GG * 4);
  float* envnum = (float*)take((size_t)GG * 4);
  float* cnt    = (float*)take((size_t)GG * 4);
  float* lcon   = (float*)take((size_t)GG * 4);
  float* na     = (float*)take((size_t)GG * 4);  // zeroed region ends before na
  float* nb     = (float*)take((size_t)GG * 4);
  float* nc     = (float*)take((size_t)GG * 4);
  long zero_floats = ((char*)na - (char*)hr) / 4;

  // -------- weight conversion (f32 -> padded, transposed f16) --------
  {
    long t1 = (long)5 * NP1 * EMBP;
    convert_pad_f16_t<<<cdiv(t1, 256), 256, 0, stream>>>(W1_g, w1g, 5, EMB, HID, EMBP, NP1);
    long t2 = (long)5 * NP2 * HIDP;
    convert_pad_f16_t<<<cdiv(t2, 256), 256, 0, stream>>>(W2_g, w2g, 5, HID, EMB, HIDP, NP2);
    long t3 = (long)2 * NP1 * EMBP;
    convert_pad_f16_t<<<cdiv(t3, 256), 256, 0, stream>>>(W1_r, w1r, 2, EMB, HID, EMBP, NP1);
    long t4 = (long)2 * NP2 * HIDP;
    convert_pad_f16_t<<<cdiv(t4, 256), 256, 0, stream>>>(W2_r, w2r, 2, HID, EMB, HIDP, NP2);
    long t5 = (long)NP1 * EMBP;
    convert_pad_f16_t<<<cdiv(t5, 256), 256, 0, stream>>>(Wg1, wg1h, 1, EMB, HID, EMBP, NP1);
  }

  // -------- node encoder --------
  encoder_kernel<<<cdiv((long)NNP * EMBP, 256), 256, 0, stream>>>(x, W_enc, b_enc, xg, xr);

  const long nCopy    = (long)NNP * EMBP;
  const long nScatter = (long)EE * (EMB / 4);

  // -------- GIN stack (encoder branch, 5 layers) --------
  for (int l = 0; l < 5; ++l) {
    copy_f32<<<cdiv(nCopy, 256), 256, 0, stream>>>(agg, xg, nCopy);
    scatter_kernel<<<cdiv(nScatter, 256), 256, 0, stream>>>(
        edge_idx, edge_attr, We_g + (size_t)l * EDIM * EMB, be_g + (size_t)l * EMB, xg, agg);
    launch_gemm(agg, EMBP, w1g + (size_t)l * NP1 * EMBP, HIDP,
                b1_g + (size_t)l * HID, HID, nullptr, h1, 1, stream);
    launch_gemm(h1, HIDP, w2g + (size_t)l * NP2 * HIDP, EMBP,
                b2_g + (size_t)l * EMB, EMB, xg, xg, (l < 4) ? 1 : 0, stream);
  }

  // -------- GIN stack (rationale branch, 2 layers) --------
  for (int l = 0; l < 2; ++l) {
    copy_f32<<<cdiv(nCopy, 256), 256, 0, stream>>>(agg, xr, nCopy);
    scatter_kernel<<<cdiv(nScatter, 256), 256, 0, stream>>>(
        edge_idx, edge_attr, We_r + (size_t)l * EDIM * EMB, be_r + (size_t)l * EMB, xr, agg);
    launch_gemm(agg, EMBP, w1r + (size_t)l * NP1 * EMBP, HIDP,
                b1_r + (size_t)l * HID, HID, nullptr, h1, 1, stream);
    launch_gemm(h1, HIDP, w2r + (size_t)l * NP2 * HIDP, EMBP,
                b2_r + (size_t)l * EMB, EMB, xr, xr, (l < 1) ? 1 : 0, stream);
  }

  // -------- gate MLP: hidden via WMMA, then logits + gumbel-softmax --------
  launch_gemm(xr, EMBP, wg1h, HIDP, bg1, HID, nullptr, h1, 1, stream);
  gate2_kernel<<<NN, 256, 0, stream>>>(h1, Wg2, bg2, gumbel, gate);

  // -------- segment reductions --------
  zero_f32<<<cdiv(zero_floats, 256), 256, 0, stream>>>(hr, zero_floats);
  count_kernel<<<cdiv((long)NN, 256), 256, 0, stream>>>(batch, cnt);
  greduce_kernel<<<cdiv((long)NN * EMB, 256), 256, 0, stream>>>(
      xg, gate, batch, hr, henv, hout, rnum, envnum);
  finalize_kernel<<<cdiv((long)GG * EMBP, 256), 256, 0, stream>>>(
      cnt, hr, henv, hout, rnum, envnum);
  norms_kernel<<<cdiv((long)GG, 256), 256, 0, stream>>>(hr, hout, henv, na, nb, nc);

  // -------- predictions + losses --------
  pred_kernel<<<GG, 256, 0, stream>>>(hr, henv, perm, Wp1, bp1, Wp2, bp2, out);
  lreg_kernel<<<1, 256, 0, stream>>>(rnum, envnum, out + 2 * GG * TASKS);
  lcon_kernel<<<GG, 256, 0, stream>>>(hr, henv, hout, na, nb, nc, lcon);
  mean_kernel<<<1, 256, 0, stream>>>(lcon, GG, out + 2 * GG * TASKS + 1);
}
